// T5AttentionBlock_86603720556880
// MI455X (gfx1250) — matrix operand order repaired and checked by reading, not compiled, not run on hardware
//
#include <hip/hip_runtime.h>
#include <cstdint>

// ---------------------------------------------------------------------------
// Problem constants (from reference)
// ---------------------------------------------------------------------------
#define N_   2
#define L_   384
#define D_   512
#define H_   8
#define HD_  64
#define DM_  2048
#define NL_  (N_ * L_)           // 768 rows
#define EPS_ 1e-5f

// ---------------------------------------------------------------------------
// CDNA5 async global->LDS copy support (guarded; falls back to sync copies)
// ---------------------------------------------------------------------------
#if defined(__HIP_DEVICE_COMPILE__) && __has_builtin(__builtin_amdgcn_global_load_async_to_lds_b128)
#define HAVE_ASYNC 1
#else
#define HAVE_ASYNC 0
#endif

// ---------------------------------------------------------------------------
// Vector types (plain ext-vectors: trivially copyable, union-safe)
// ---------------------------------------------------------------------------
typedef __attribute__((ext_vector_type(16))) __bf16  v16bf;
typedef __attribute__((ext_vector_type(8)))  float   v8f;
typedef __attribute__((ext_vector_type(4))) unsigned u32x4;
typedef __attribute__((ext_vector_type(4)))  float   f32x4;

// GCC-style int4 vector: exact pointee type of the async-load builtin params
typedef int gcc_v4i __attribute__((vector_size(16)));
#if HAVE_ASYNC
typedef __attribute__((address_space(1))) gcc_v4i as1_v4i;   // global (__device__)
typedef __attribute__((address_space(3))) gcc_v4i as3_v4i;   // LDS (__shared__)
#endif

union Frag16 { v16bf v; u32x4 u4[2]; };   // 32B operand fragment (A or B)

__device__ __forceinline__ unsigned short bfbits(float f) {
    unsigned u = __float_as_uint(f);
    unsigned r = u + 0x7FFFu + ((u >> 16) & 1u);   // round-to-nearest-even
    return (unsigned short)(r >> 16);
}
__device__ __forceinline__ __bf16 f2bf(float f) {
    unsigned short s = bfbits(f);
    __bf16 o; __builtin_memcpy(&o, &s, 2); return o;
}
__device__ __forceinline__ float wave_sum(float v) {
#pragma unroll
    for (int m = 16; m; m >>= 1) v += __shfl_xor(v, m, 32);
    return v;
}
__device__ __forceinline__ float wave_max(float v) {
#pragma unroll
    for (int m = 16; m; m >>= 1) v = fmaxf(v, __shfl_xor(v, m, 32));
    return v;
}
__device__ __forceinline__ v8f vzero8() {
    v8f z;
#pragma unroll
    for (int i = 0; i < 8; ++i) z[i] = 0.f;
    return z;
}

// 16-byte global -> LDS copy (async DMA when available; ASYNCcnt-tracked)
__device__ __forceinline__ void async16(const __bf16* g, __bf16* l) {
#if HAVE_ASYNC
    __builtin_amdgcn_global_load_async_to_lds_b128(
        (as1_v4i*)(unsigned long long)(uintptr_t)g,
        (as3_v4i*)(unsigned)(uintptr_t)l,
        0, 0);
#else
    *reinterpret_cast<u32x4*>(l) = *reinterpret_cast<const u32x4*>(g);
#endif
}
__device__ __forceinline__ void wait_async0() {
#if HAVE_ASYNC
#if __has_builtin(__builtin_amdgcn_s_wait_asynccnt)
    __builtin_amdgcn_s_wait_asynccnt(0);
#else
    asm volatile("s_wait_asynccnt 0x0" ::: "memory");
#endif
#endif
}
__device__ __forceinline__ void wait_async3() {   // allow 3 in-flight (next tile)
#if HAVE_ASYNC
#if __has_builtin(__builtin_amdgcn_s_wait_asynccnt)
    __builtin_amdgcn_s_wait_asynccnt(3);
#else
    asm volatile("s_wait_asynccnt 0x3" ::: "memory");
#endif
#endif
}

// ---------------------------------------------------------------------------
// One-shot weight conversion f32 -> bf16 (row-major preserved)
// ---------------------------------------------------------------------------
__global__ __launch_bounds__(256)
void k_castw(const float* __restrict__ src, __bf16* __restrict__ dst, int n) {
    const int i = (blockIdx.x * 256 + threadIdx.x) * 8;
    if (i >= n) return;
    const f32x4 a = *reinterpret_cast<const f32x4*>(src + i);
    const f32x4 b = *reinterpret_cast<const f32x4*>(src + i + 4);
    u32x4 pk;
    pk.x = (unsigned)bfbits(a.x) | ((unsigned)bfbits(a.y) << 16);
    pk.y = (unsigned)bfbits(a.z) | ((unsigned)bfbits(a.w) << 16);
    pk.z = (unsigned)bfbits(b.x) | ((unsigned)bfbits(b.y) << 16);
    pk.w = (unsigned)bfbits(b.z) | ((unsigned)bfbits(b.w) << 16);
    *reinterpret_cast<u32x4*>(dst + i) = pk;
}

// ---------------------------------------------------------------------------
// LayerNorm (f32 in -> bf16 out), one block (256 thr) per row of 512
// ---------------------------------------------------------------------------
__global__ __launch_bounds__(256)
void k_layernorm(const float* __restrict__ in, const float* __restrict__ g,
                 const float* __restrict__ b, __bf16* __restrict__ out) {
    const int row = blockIdx.x, t = threadIdx.x;
    const int wave = t >> 5, lane = t & 31;
    const float* r = in + (size_t)row * D_;
    const float a0 = r[t], a1 = r[t + 256];

    __shared__ float red[8];
    float s = wave_sum(a0 + a1);
    if (lane == 0) red[wave] = s;
    __syncthreads();
    float mean = 0.f;
#pragma unroll
    for (int i = 0; i < 8; ++i) mean += red[i];
    mean *= (1.0f / D_);
    __syncthreads();

    const float d0 = a0 - mean, d1 = a1 - mean;
    s = wave_sum(d0 * d0 + d1 * d1);
    if (lane == 0) red[wave] = s;
    __syncthreads();
    float var = 0.f;
#pragma unroll
    for (int i = 0; i < 8; ++i) var += red[i];
    var *= (1.0f / D_);

    const float rstd = rsqrtf(var + EPS_);
    out[(size_t)row * D_ + t]       = f2bf(d0 * rstd * g[t] + b[t]);
    out[(size_t)row * D_ + t + 256] = f2bf(d1 * rstd * g[t + 256] + b[t + 256]);
}

// ---------------------------------------------------------------------------
// WMMA GEMM:  C[m,n] = sum_k A[m,k] * W[n,k] + bvec[n], epilogue by mode.
//   A: bf16 (M x K) row-major.  W: bf16 (Ncols x K) row-major (pre-converted).
//   Block tile 128x64, 256 threads = 8 waves, each wave owns a 16x64 strip
//   (4 v_wmma_f32_16x16x32_bf16 accumulators).  LDS tiles are double-buffered
//   and filled with global_load_async_to_lds_b128 (ASYNCcnt), overlapping the
//   next tile's DMA with the current tile's WMMAs.
// ---------------------------------------------------------------------------
enum { MODE_Q = 0, MODE_KV = 1, MODE_PROJ = 2, MODE_SILU = 3, MODE_FINAL = 4 };

__global__ __launch_bounds__(256)
void k_gemm(const __bf16* __restrict__ A, const __bf16* __restrict__ W,
            const float* __restrict__ bvec, int Kdim, int mode,
            void* __restrict__ out0, void* __restrict__ out1,
            const float* __restrict__ resid) {
    const int t = threadIdx.x, wave = t >> 5, lane = t & 31;
    const int col0 = blockIdx.x * 64, row0 = blockIdx.y * 128;
    const int Ncols = gridDim.x * 64;

    __shared__ __bf16 sA[2][128 * 32];   // 2 x 8 KB
    __shared__ __bf16 sW[2][64 * 32];    // 2 x 4 KB

    v8f acc[4];
#pragma unroll
    for (int i = 0; i < 4; ++i) acc[i] = vzero8();

    const int am = t >> 1, ah = (t & 1) * 16;   // A staging: 32B/thread
    const int wn = t >> 2, wq = (t & 3) * 8;    // W staging: 16B/thread
    const __bf16* gA = A + (size_t)(row0 + am) * Kdim + ah;
    const __bf16* gW = W + (size_t)(col0 + wn) * Kdim + wq;

    auto stage = [&](int ik, int buf) {
        const int k0 = ik * 32;
        async16(gA + k0,     &sA[buf][am * 32 + ah]);
        async16(gA + k0 + 8, &sA[buf][am * 32 + ah + 8]);
        async16(gW + k0,     &sW[buf][wn * 32 + wq]);
    };

    const int nk = Kdim >> 5;
    stage(0, 0);
    for (int ik = 0; ik < nk; ++ik) {
        const int cur = ik & 1;
        if (ik + 1 < nk) { stage(ik + 1, cur ^ 1); wait_async3(); }
        else             { wait_async0(); }
        __syncthreads();

        // A fragment: 16x32, lane m = lane&15, two 8-elem runs
        const int fm  = wave * 16 + (lane & 15);
        const int off = (lane >> 4) * 8;
        Frag16 af;
        af.u4[0] = *reinterpret_cast<const u32x4*>(&sA[cur][fm * 32 + off]);
        af.u4[1] = *reinterpret_cast<const u32x4*>(&sA[cur][fm * 32 + off + 16]);

        const int bk = (lane >> 4) * 8;    // B fragment: two 8-elem runs
#pragma unroll
        for (int nt = 0; nt < 4; ++nt) {
            Frag16 bf;
            const __bf16* bp = &sW[cur][(nt * 16 + (lane & 15)) * 32 + bk];
            bf.u4[0] = ((const u32x4*)bp)[0];
            bf.u4[1] = ((const u32x4*)bp)[2];
            acc[nt] = __builtin_amdgcn_wmma_f32_16x16x32_bf16(
                false, af.v, false, bf.v, (short)0, acc[nt], false, false);
        }
        __syncthreads();
    }

    // ---- epilogue ----
    const int mrow = row0 + wave * 16;
#pragma unroll
    for (int nt = 0; nt < 4; ++nt) {
        const int col = col0 + nt * 16 + (lane & 15);
        const float bb = bvec[col];
#pragma unroll
        for (int r = 0; r < 8; ++r) {
            const int row = mrow + r + (lane >> 4) * 8;
            const float val = acc[nt][r] + bb;
            if (mode == MODE_Q) {
                const int nn = row / L_, ll = row % L_;
                const int h = col >> 6, hd = col & 63;
                ((__bf16*)out0)[((size_t)(nn * H_ + h) * L_ + ll) * HD_ + hd] = f2bf(val);
            } else if (mode == MODE_KV) {
                const int nn = row / L_, ll = row % L_;
                const int sel = col >> 9, c = col & 511;
                const int h = c >> 6, hd = c & 63;
                if (sel == 0)   // K: per-head row-major
                    ((__bf16*)out0)[((size_t)(nn * H_ + h) * L_ + ll) * HD_ + hd] = f2bf(val);
                else            // V: stored TRANSPOSED [n,h,hd,l] for PV WMMA
                    ((__bf16*)out1)[((size_t)(nn * H_ + h) * HD_ + hd) * L_ + ll] = f2bf(val);
            } else if (mode == MODE_PROJ || mode == MODE_FINAL) {
                const size_t idx = (size_t)row * D_ + col;
                ((float*)out0)[idx] = resid[idx] + val;
            } else {            // MODE_SILU
                const float sv = val / (1.f + __expf(-val));
                ((__bf16*)out0)[(size_t)row * Ncols + col] = f2bf(sv);
            }
        }
    }
}

// ---------------------------------------------------------------------------
// Relative-position bias: bias[n,h,i,j] = pd[n,i,j,:] . edw[h,:] + edb[h]
// HBM-bound (604 MB stream). One wave per (n,i,j) row; edw cached in LDS.
// ---------------------------------------------------------------------------
__global__ __launch_bounds__(256)
void k_bias(const float* __restrict__ pd, const float* __restrict__ edw,
            const float* __restrict__ edb, float* __restrict__ biasbuf) {
    const int t = threadIdx.x, wave = t >> 5, lane = t & 31;
    __shared__ float sE[H_][D_];   // 16 KB
    for (int i = t; i < H_ * D_; i += 256) sE[i >> 9][i & 511] = edw[i];
    __syncthreads();

    const int row = blockIdx.x * 8 + wave;             // flat (n,i,j)
    const float* prow = pd + (size_t)row * D_;
    __builtin_prefetch((const void*)(prow + 8 * D_), 0, 2);   // next row group
    float acc[H_];
#pragma unroll
    for (int h = 0; h < H_; ++h) acc[h] = 0.f;

    for (int e0 = lane * 4; e0 < D_; e0 += 128) {
        const f32x4 pv = *reinterpret_cast<const f32x4*>(prow + e0);
#pragma unroll
        for (int h = 0; h < H_; ++h)
            acc[h] += pv.x * sE[h][e0] + pv.y * sE[h][e0 + 1] +
                      pv.z * sE[h][e0 + 2] + pv.w * sE[h][e0 + 3];
    }
#pragma unroll
    for (int h = 0; h < H_; ++h) acc[h] = wave_sum(acc[h]);
    if (lane == 0) {
        const int n = row / (L_ * L_), rem = row % (L_ * L_);
        const int i = rem / L_, j = rem % L_;
#pragma unroll
        for (int h = 0; h < H_; ++h)
            biasbuf[((size_t)((n * H_ + h) * L_) + i) * L_ + j] = acc[h] + edb[h];
    }
}

// ---------------------------------------------------------------------------
// Attention: one block per (n, h, 16-row q tile).
//   S = (Q Kt)*scale + bias + mask -> softmax -> O = P V   (all via WMMA)
// ---------------------------------------------------------------------------
__global__ __launch_bounds__(256)
void k_attn(const __bf16* __restrict__ q, const __bf16* __restrict__ kmat,
            const __bf16* __restrict__ vT, const float* __restrict__ biasbuf,
            const unsigned char* __restrict__ padmask, __bf16* __restrict__ o) {
    const int t = threadIdx.x, wave = t >> 5, lane = t & 31;
    const int ntile = L_ / 16;                       // 24
    const int it = blockIdx.x % ntile;
    const int h  = (blockIdx.x / ntile) % H_;
    const int n  = blockIdx.x / (ntile * H_);
    const int i0 = it * 16;
    const size_t headL = (size_t)(n * H_ + h) * L_;  // q/k row base (x HD_)

    __shared__ float  sS[16][L_];       // 24 KB scores / probs (f32)
    __shared__ __bf16 sP[16][L_];       // 12 KB probs (bf16 A-operand)
    __shared__ float  sO[8][256];       // 8 KB partial O per wave
    __shared__ float  sMaskAdd[L_];     // 1.5 KB additive mask
    __shared__ int    sAnyValid;

    if (t == 0) sAnyValid = 0;
    __syncthreads();
    int anyv = 0;
    for (int j = t; j < L_; j += 256) if (!padmask[n * L_ + j]) anyv = 1;
    if (anyv) atomicOr(&sAnyValid, 1);
    __syncthreads();
    const bool allpad = (sAnyValid == 0);
    for (int j = t; j < L_; j += 256)
        sMaskAdd[j] = (allpad || !padmask[n * L_ + j]) ? 0.f : -1e30f;
    __builtin_prefetch((const void*)(biasbuf + (headL + i0 + wave * 2) * L_), 0, 3);

    // Q fragments (16 x 64, kept in registers for the whole tile)
    const int foff = (lane >> 4) * 8;
    const __bf16* qrow = q + (headL + i0 + (lane & 15)) * HD_;
    Frag16 qa[2];
#pragma unroll
    for (int ks = 0; ks < 2; ++ks) {
        qa[ks].u4[0] = *reinterpret_cast<const u32x4*>(qrow + ks * 32 + foff);
        qa[ks].u4[1] = *reinterpret_cast<const u32x4*>(qrow + ks * 32 + foff + 16);
    }

    // ---- scores: each wave does 3 j-tiles of 16 ----
#pragma unroll
    for (int jl = 0; jl < 3; ++jl) {
        const int j0 = (wave * 3 + jl) * 16;
        v8f acc = vzero8();
        const __bf16* krow = kmat + (headL + j0 + (lane & 15)) * HD_;
#pragma unroll
        for (int ks = 0; ks < 2; ++ks) {
            Frag16 kb;
            const int doff = ks * 32 + (lane >> 4) * 8;
            kb.u4[0] = *reinterpret_cast<const u32x4*>(krow + doff);
            kb.u4[1] = *reinterpret_cast<const u32x4*>(krow + doff + 16);
            acc = __builtin_amdgcn_wmma_f32_16x16x32_bf16(
                false, qa[ks].v, false, kb.v, (short)0, acc, false, false);
        }
#pragma unroll
        for (int r = 0; r < 8; ++r)
            sS[r + (lane >> 4) * 8][j0 + (lane & 15)] = acc[r];
    }
    __syncthreads();

    // ---- softmax: 2 rows per wave ----
    const float scale = 0.125f;   // HD^-0.5
#pragma unroll
    for (int rr = 0; rr < 2; ++rr) {
        const int i = wave * 2 + rr;
        const float* brow = biasbuf + (headL + i0 + i) * L_;
        float mx = -3.0e38f;
        for (int j = lane; j < L_; j += 32) {
            const float v = sS[i][j] * scale + brow[j] + sMaskAdd[j];
            sS[i][j] = v;
            mx = fmaxf(mx, v);
        }
        mx = wave_max(mx);
        float sum = 0.f;
        for (int j = lane; j < L_; j += 32) {
            const float e = __expf(sS[i][j] - mx);
            sS[i][j] = e;
            sum += e;
        }
        sum = wave_sum(sum);
        const float inv = 1.0f / sum;
        for (int j = lane; j < L_; j += 32)
            sP[i][j] = f2bf(sS[i][j] * inv);
    }
    __syncthreads();

    // ---- O = P @ V: wave pairs split K (j) dimension, 4 d-tiles ----
    const int dt = wave & 3, kh = wave >> 2;
    v8f oacc = vzero8();
    const __bf16* vrow = vT + ((size_t)((n * H_ + h) * HD_) + dt * 16 + (lane & 15)) * L_;
#pragma unroll
    for (int s = 0; s < 6; ++s) {
        const int j0 = kh * 192 + s * 32;
        Frag16 pa, vb;
        pa.u4[0] = *reinterpret_cast<const u32x4*>(&sP[lane & 15][j0 + foff]);
        pa.u4[1] = *reinterpret_cast<const u32x4*>(&sP[lane & 15][j0 + foff + 16]);
        const int jo = j0 + (lane >> 4) * 8;
        vb.u4[0] = *reinterpret_cast<const u32x4*>(vrow + jo);
        vb.u4[1] = *reinterpret_cast<const u32x4*>(vrow + jo + 16);
        oacc = __builtin_amdgcn_wmma_f32_16x16x32_bf16(
            false, pa.v, false, vb.v, (short)0, oacc, false, false);
    }
#pragma unroll
    for (int r = 0; r < 8; ++r) sO[wave][r * 32 + lane] = oacc[r];
    __syncthreads();

    if (wave < 4) {
#pragma unroll
        for (int r = 0; r < 8; ++r) {
            const float val = sO[wave][r * 32 + lane] + sO[wave + 4][r * 32 + lane];
            const int i  = r + (lane >> 4) * 8;
            const int hd = dt * 16 + (lane & 15);
            o[((size_t)(n * L_) + i0 + i) * D_ + h * HD_ + hd] =
                f2bf(allpad ? 0.f : val);
        }
    }
}

// ---------------------------------------------------------------------------
// Host orchestration
// ---------------------------------------------------------------------------
extern "C" void kernel_launch(void* const* d_in, const int* in_sizes, int n_in,
                              void* d_out, int out_size, void* d_ws, size_t ws_size,
                              hipStream_t stream) {
    (void)in_sizes; (void)n_in; (void)out_size; (void)ws_size;
    const float* x    = (const float*)d_in[0];
    const float* mem  = (const float*)d_in[1];
    const float* pd   = (const float*)d_in[2];
    const unsigned char* pmask = (const unsigned char*)d_in[3];
    const float* qw   = (const float*)d_in[4];
    const float* qb   = (const float*)d_in[5];
    const float* kvw  = (const float*)d_in[6];
    const float* kvb  = (const float*)d_in[7];
    const float* pw   = (const float*)d_in[8];
    const float* pb   = (const float*)d_in[9];
    const float* edw  = (const float*)d_in[10];
    const float* edb  = (const float*)d_in[11];
    const float* ln1g = (const float*)d_in[12];
    const float* ln1b = (const float*)d_in[13];
    const float* ln2g = (const float*)d_in[14];
    const float* ln2b = (const float*)d_in[15];
    const float* w1   = (const float*)d_in[16];
    const float* b1   = (const float*)d_in[17];
    const float* w2   = (const float*)d_in[18];
    const float* b2   = (const float*)d_in[19];
    const float* w3   = (const float*)d_in[20];
    const float* b3   = (const float*)d_in[21];

    char* ws = (char*)d_ws;
    size_t off = 0;
    auto alloc = [&](size_t bytes) -> char* {
        char* p = ws + off;
        off += (bytes + 255) & ~(size_t)255;
        return p;
    };
    __bf16* aX   = (__bf16*)alloc((size_t)NL_ * D_ * 2);
    __bf16* aM   = (__bf16*)alloc((size_t)NL_ * D_ * 2);
    __bf16* qh   = (__bf16*)alloc((size_t)NL_ * D_ * 2);
    __bf16* kh   = (__bf16*)alloc((size_t)NL_ * D_ * 2);
    __bf16* vTh  = (__bf16*)alloc((size_t)NL_ * D_ * 2);
    __bf16* ob   = (__bf16*)alloc((size_t)NL_ * D_ * 2);
    __bf16* hln  = (__bf16*)alloc((size_t)NL_ * D_ * 2);
    __bf16* h1   = (__bf16*)alloc((size_t)NL_ * DM_ * 2);
    __bf16* h2   = (__bf16*)alloc((size_t)NL_ * DM_ * 2);
    float*  x1   = (float*)alloc((size_t)NL_ * D_ * 4);
    float*  bbuf = (float*)alloc((size_t)N_ * H_ * L_ * L_ * 4);
    // bf16 weights
    __bf16* qwB  = (__bf16*)alloc((size_t)D_ * D_ * 2);
    __bf16* kvwB = (__bf16*)alloc((size_t)2 * D_ * D_ * 2);
    __bf16* pwB  = (__bf16*)alloc((size_t)D_ * D_ * 2);
    __bf16* w1B  = (__bf16*)alloc((size_t)DM_ * D_ * 2);
    __bf16* w2B  = (__bf16*)alloc((size_t)DM_ * DM_ * 2);
    __bf16* w3B  = (__bf16*)alloc((size_t)D_ * DM_ * 2);

    // 0) weight conversion (cheap: ~29MB read)
    k_castw<<<(D_ * D_) / 2048,      256, 0, stream>>>(qw,  qwB,  D_ * D_);
    k_castw<<<(2 * D_ * D_) / 2048,  256, 0, stream>>>(kvw, kvwB, 2 * D_ * D_);
    k_castw<<<(D_ * D_) / 2048,      256, 0, stream>>>(pw,  pwB,  D_ * D_);
    k_castw<<<(DM_ * D_) / 2048,     256, 0, stream>>>(w1,  w1B,  DM_ * D_);
    k_castw<<<(DM_ * DM_) / 2048,    256, 0, stream>>>(w2,  w2B,  DM_ * DM_);
    k_castw<<<(D_ * DM_) / 2048,     256, 0, stream>>>(w3,  w3B,  D_ * DM_);

    // 1) LayerNorm of x and mem
    k_layernorm<<<NL_, 256, 0, stream>>>(x,   ln1g, ln1b, aX);
    k_layernorm<<<NL_, 256, 0, stream>>>(mem, ln1g, ln1b, aM);

    // 2) Q = aX @ qw^T + qb  -> per-head [n,h,l,hd]
    k_gemm<<<dim3(D_ / 64, NL_ / 128), 256, 0, stream>>>(
        aX, qwB, qb, D_, MODE_Q, qh, nullptr, nullptr);

    // 3) KV = aM @ kvw^T + kvb -> K per-head, V transposed per-head
    k_gemm<<<dim3((2 * D_) / 64, NL_ / 128), 256, 0, stream>>>(
        aM, kvwB, kvb, D_, MODE_KV, kh, vTh, nullptr);

    // 4) relative-position bias (streams the 604MB pairwise tensor once)
    k_bias<<<(N_ * L_ * L_) / 8, 256, 0, stream>>>(pd, edw, edb, bbuf);

    // 5) attention
    k_attn<<<N_ * H_ * (L_ / 16), 256, 0, stream>>>(qh, kh, vTh, bbuf, pmask, ob);

    // 6) x1 = x + out @ pw^T + pb
    k_gemm<<<dim3(D_ / 64, NL_ / 128), 256, 0, stream>>>(
        ob, pwB, pb, D_, MODE_PROJ, x1, nullptr, x);

    // 7) LayerNorm2
    k_layernorm<<<NL_, 256, 0, stream>>>(x1, ln2g, ln2b, hln);

    // 8) h1 = silu(hln @ w1^T + b1)
    k_gemm<<<dim3(DM_ / 64, NL_ / 128), 256, 0, stream>>>(
        hln, w1B, b1, D_, MODE_SILU, h1, nullptr, nullptr);

    // 9) h2 = silu(h1 @ w2^T + b2)
    k_gemm<<<dim3(DM_ / 64, NL_ / 128), 256, 0, stream>>>(
        h1, w2B, b2, DM_, MODE_SILU, h2, nullptr, nullptr);

    // 10) out = x1 + h2 @ w3^T + b3
    k_gemm<<<dim3(D_ / 64, NL_ / 128), 256, 0, stream>>>(
        h2, w3B, b3, DM_, MODE_FINAL, d_out, nullptr, x1);
}
